// StatefulRNNLayer_44693429682389
// MI455X (gfx1250) — matrix operand order repaired and checked by reading, not compiled, not run on hardware
//
#include <hip/hip_runtime.h>
#include <hip/hip_bf16.h>
#include <cstdint>

// ---------------------------------------------------------------------------
// Stateful RNN: xproj = x@Wx + b  (WMMA GEMM, TDM-parked B tile), then scan
//   h_t = tanh(xproj_t + h_{t-1} @ Wh)  persistent kernel, TDM-parked Wh slice
// ---------------------------------------------------------------------------

typedef __bf16 bf16x16 __attribute__((ext_vector_type(16)));
typedef __bf16 bf16x8  __attribute__((ext_vector_type(8)));
typedef float  f32x8   __attribute__((ext_vector_type(8)));
typedef unsigned u32x4 __attribute__((ext_vector_type(4)));
typedef unsigned u32x8 __attribute__((ext_vector_type(8)));

#define B_   64
#define S_   512
#define DIN  512
#define DH   1024
#define MTOT (B_ * S_)      // 32768 rows of the xproj GEMM

// workspace layout (bytes)
#define WXT_OFF  0                                   // bf16 [DH][DIN]  (N-major, transposed)
#define WHT_OFF  (WXT_OFF + DH * DIN * 2)            // bf16 [DH][DH]
#define HB0_OFF  (WHT_OFF + DH * DH * 2)             // bf16 [B_][DH]
#define HB1_OFF  (HB0_OFF + B_ * DH * 2)             // bf16 [B_][DH]
#define CTR_OFF  (HB1_OFF + B_ * DH * 2)             // unsigned counter

__device__ __forceinline__ unsigned short f2bf(float f) {
  unsigned u = __float_as_uint(f);
  unsigned r = ((u >> 16) & 1u) + 0x7FFFu;   // round-to-nearest-even
  return (unsigned short)((u + r) >> 16);
}

// ---------------------------------------------------------------------------
// Tensor Data Mover: 2D bf16 tile load, full-tile (tile == tensor dims),
// with LDS padding of 4 DWORDs (16B) after every 256 DWORDs (1024B) so the
// effective LDS row stride is (row_bytes + 16*(row_bytes/1024)) — bank-spread.
// D# layout per CDNA5 ISA.
// ---------------------------------------------------------------------------
__device__ __forceinline__ void tdm_load_2d_bf16(unsigned lds_off, const void* gptr,
                                                 unsigned dim0, unsigned dim1,
                                                 unsigned stride0) {
  const unsigned long long ga = (unsigned long long)gptr;
  u32x4 g0;
  g0[0] = 1u;                                                   // count=1, no gather
  g0[1] = lds_off;                                              // lds_addr (bytes)
  g0[2] = (unsigned)ga;                                         // global_addr[31:0]
  g0[3] = ((unsigned)(ga >> 32) & 0x01FFFFFFu) | (2u << 30);    // addr[56:32] | type=2
  u32x8 g1;
  g1[0] = (1u << 16)      // data_size = 2 bytes
        | (1u << 20)      // pad_enable
        | (7u << 22)      // pad_interval: 256 DWORDs (1024 B)
        | (3u << 25);     // pad_amount: 4 DWORDs (16 B)
  g1[1] = (dim0 & 0xFFFFu) << 16;                               // tensor_dim0 lo16
  g1[2] = ((dim0 >> 16) & 0xFFFFu) | ((dim1 & 0xFFFFu) << 16);  // dim0 hi16 | dim1 lo16
  g1[3] = ((dim1 >> 16) & 0xFFFFu) | ((dim0 & 0xFFFFu) << 16);  // dim1 hi16 | tile_dim0
  g1[4] = dim1 & 0xFFFFu;                                       // tile_dim1 (tile_dim2=0)
  g1[5] = stride0;                                              // tensor_dim0_stride lo32
  g1[6] = 0u;
  g1[7] = 0u;
  asm volatile("tensor_load_to_lds %0, %1" :: "s"(g0), "s"(g1) : "memory");
}

// ---------------------------------------------------------------------------
// Kernel 0: reset barrier counter, convert h0 -> bf16, convert+transpose W
// ---------------------------------------------------------------------------
__global__ void rnn_init(const float* __restrict__ W, const float* __restrict__ h0,
                         unsigned short* __restrict__ wxT, unsigned short* __restrict__ whT,
                         unsigned short* __restrict__ hb0, unsigned* __restrict__ ctr) {
  const int idx = blockIdx.x * blockDim.x + threadIdx.x;
  const int stride = gridDim.x * blockDim.x;
  if (idx == 0) *ctr = 0u;
  for (int i = idx; i < (DIN + DH) * DH; i += stride) {
    const int k = i / DH;
    const int n = i % DH;
    const unsigned short v = f2bf(W[i]);
    if (k < DIN) wxT[n * DIN + k]        = v;   // WxT[n][k] = Wx[k][n]
    else         whT[n * DH + (k - DIN)] = v;   // WhT[n][k] = Wh[k][n]
  }
  for (int i = idx; i < B_ * DH; i += stride) hb0[i] = f2bf(h0[i]);
}

// ---------------------------------------------------------------------------
// Kernel 1: xproj GEMM  out[M][DH] = x[M][DIN] @ Wx + b     (M = B*S)
// B tile (128 cols x 512 K, bf16) parked in LDS once via TDM; only the
// fp32->bf16 A tile is staged per K step. 8 waves, 8 WMMA accs each.
// ---------------------------------------------------------------------------
#define BM   128
#define BN   128
#define BK   32
#define ASTR 40    // halves; 80B row stride, 16B-aligned chunks, conflict-free
#define BSTR 520   // halves; TDM-padded stride: 1024B row + 16B pad

__global__ __launch_bounds__(256)
void xproj_gemm(const float* __restrict__ x, const unsigned short* __restrict__ wxT,
                const float* __restrict__ bias, float* __restrict__ out) {
  __shared__ unsigned short As[BM * ASTR];
  __shared__ unsigned short Bs[BN * BSTR];

  const int tid  = threadIdx.x;
  const int lane = tid & 31;
  const int wave = tid >> 5;
  const int wm   = wave & 3;          // wave row group (32 rows)
  const int wn   = wave >> 2;         // wave col group (64 cols)
  const int hi   = (lane >> 4) & 1;   // half-wave select
  const int l16  = lane & 15;

  const int m0 = blockIdx.x * BM;
  const int n0 = blockIdx.y * BN;

  // park the whole B tile (WxT rows n0..n0+127, all 512 K) via TDM
  if (wave == 0) {
    tdm_load_2d_bf16((unsigned)(uintptr_t)&Bs[0], wxT + (size_t)n0 * DIN,
                     DIN, BN, DIN);
    __builtin_amdgcn_s_wait_tensorcnt(0);
  }

  const f32x8 zero = {};
  f32x8 acc[2][4];
#pragma unroll
  for (int mt = 0; mt < 2; ++mt)
#pragma unroll
    for (int nt = 0; nt < 4; ++nt) acc[mt][nt] = zero;

  for (int kb = 0; kb < DIN; kb += BK) {
    // --- stage A tile (128x32 fp32 -> bf16 in LDS) ---
#pragma unroll
    for (int i = 0; i < 4; ++i) {
      const int t  = tid + i * 256;         // float4 unit, 0..1023
      const int r  = t >> 3;
      const int c4 = t & 7;
      const float4 v = *(const float4*)(x + (size_t)(m0 + r) * DIN + kb + c4 * 4);
      unsigned short* dst = &As[r * ASTR + c4 * 4];
      dst[0] = f2bf(v.x); dst[1] = f2bf(v.y); dst[2] = f2bf(v.z); dst[3] = f2bf(v.w);
    }
    __syncthreads();

    // --- fragments: contiguous k order (k = hi*16 + s), same for A and B ---
    union F { bf16x16 v; bf16x8 h[2]; } af[2], bf[4];
#pragma unroll
    for (int mt = 0; mt < 2; ++mt) {
      const unsigned short* ap = &As[(wm * 32 + mt * 16 + l16) * ASTR + hi * 16];
      af[mt].h[0] = *(const bf16x8*)(ap);        // K = hi*16 .. hi*16+7
      af[mt].h[1] = *(const bf16x8*)(ap + 8);    // K = hi*16+8 ..
    }
#pragma unroll
    for (int nt = 0; nt < 4; ++nt) {
      const unsigned short* bp = &Bs[(wn * 64 + nt * 16 + l16) * BSTR + kb + hi * 16];
      bf[nt].h[0] = *(const bf16x8*)(bp);        // K = kb + hi*16 ..
      bf[nt].h[1] = *(const bf16x8*)(bp + 8);
    }
#pragma unroll
    for (int mt = 0; mt < 2; ++mt)
#pragma unroll
      for (int nt = 0; nt < 4; ++nt)
        acc[mt][nt] = __builtin_amdgcn_wmma_f32_16x16x32_bf16(
            false, af[mt].v, false, bf[nt].v, (short)0, acc[mt][nt], false, false);
    __syncthreads();
  }

  // --- epilogue: + bias, store fp32 xproj into d_out ---
#pragma unroll
  for (int nt = 0; nt < 4; ++nt) {
    const int col = n0 + wn * 64 + nt * 16 + l16;
    const float bv = bias[col];
#pragma unroll
    for (int mt = 0; mt < 2; ++mt) {
#pragma unroll
      for (int v = 0; v < 8; ++v) {
        const int row = m0 + wm * 32 + mt * 16 + hi * 8 + v;
        out[(size_t)row * DH + col] = acc[mt][nt][v] + bv;
      }
    }
  }
}

// ---------------------------------------------------------------------------
// Kernel 2: persistent recurrent scan.
// 16 workgroups, each owns 64 output columns; its 1024x64 slice of WhT is
// parked in LDS once via TDM (130 KB incl. pad) for all 512 steps.
// TDM inserts 16B pad every 1024B -> row stride 2080B, mid-row pad at K=512.
// ---------------------------------------------------------------------------
#define NWG   16
#define WHSTR 1040   // halves per LDS row (2048B data + 2x16B pad)

__global__ __launch_bounds__(256)
void rnn_scan(const unsigned short* __restrict__ whT,
              float* __restrict__ out,          // (B,S,DH): xproj in, h out
              float* __restrict__ hlast,        // (B,DH)
              unsigned short* __restrict__ hb0,
              unsigned short* __restrict__ hb1,
              unsigned* __restrict__ ctr) {
  __shared__ unsigned short Wl[64 * WHSTR];

  const int tid  = threadIdx.x;
  const int lane = tid & 31;
  const int wave = tid >> 5;
  const int wm   = wave & 3;          // 16 batch rows per wave group
  const int wn   = wave >> 2;         // 32 cols per wave group
  const int hi   = (lane >> 4) & 1;
  const int l16  = lane & 15;
  const int n0   = blockIdx.x * 64;

  // park Wh slice in LDS once via TDM (reused 512 times)
  if (wave == 0) {
    tdm_load_2d_bf16((unsigned)(uintptr_t)&Wl[0], whT + (size_t)n0 * DH,
                     DH, 64, DH);
    __builtin_amdgcn_s_wait_tensorcnt(0);
  }
  __syncthreads();

  unsigned short* hcur = hb0;
  unsigned short* hnxt = hb1;
  unsigned target = NWG;

  for (int t = 0; t < S_; ++t) {
    f32x8 acc0 = {}, acc1 = {};
    const int arow = wm * 16 + l16;                      // batch index for A rows
    const unsigned short* hrow = hcur + (size_t)arow * DH;
    const unsigned short* b0base = &Wl[(wn * 32 + l16) * WHSTR + hi * 16];
    const unsigned short* b1base = &Wl[(wn * 32 + 16 + l16) * WHSTR + hi * 16];

#pragma unroll 8
    for (int kb = 0; kb < DH; kb += 32) {
      const int kadj = kb + ((kb >> 9) << 3);   // +8 halves past mid-row TDM pad
      union F { bf16x16 v; bf16x8 h[2]; } af, b0, b1;
      af.h[0] = *(const bf16x8*)(hrow + kb + hi * 16);
      af.h[1] = *(const bf16x8*)(hrow + kb + hi * 16 + 8);
      b0.h[0] = *(const bf16x8*)(b0base + kadj);
      b0.h[1] = *(const bf16x8*)(b0base + kadj + 8);
      b1.h[0] = *(const bf16x8*)(b1base + kadj);
      b1.h[1] = *(const bf16x8*)(b1base + kadj + 8);
      acc0 = __builtin_amdgcn_wmma_f32_16x16x32_bf16(
          false, af.v, false, b0.v, (short)0, acc0, false, false);
      acc1 = __builtin_amdgcn_wmma_f32_16x16x32_bf16(
          false, af.v, false, b1.v, (short)0, acc1, false, false);
    }

    // h_new = tanh(xproj + h@Wh); publish fp32 to out, bf16 to next h buffer
#pragma unroll
    for (int nt = 0; nt < 2; ++nt) {
      const f32x8 a = nt ? acc1 : acc0;
      const int col = n0 + wn * 32 + nt * 16 + l16;
#pragma unroll
      for (int v = 0; v < 8; ++v) {
        const int b = wm * 16 + hi * 8 + v;
        const size_t oidx = ((size_t)b * S_ + t) * DH + col;
        const float val = tanhf(out[oidx] + a[v]);
        out[oidx] = val;
        hnxt[b * DH + col] = f2bf(val);
        if (t == S_ - 1) hlast[b * DH + col] = val;
      }
    }

    // ---- device-wide barrier (monotonic counter, release/acquire) ----
    __threadfence();
    __syncthreads();
    if (tid == 0) {
      __hip_atomic_fetch_add(ctr, 1u, __ATOMIC_RELEASE, __HIP_MEMORY_SCOPE_AGENT);
      while (__hip_atomic_load(ctr, __ATOMIC_ACQUIRE, __HIP_MEMORY_SCOPE_AGENT) < target)
        __builtin_amdgcn_s_sleep(1);
    }
    __syncthreads();
    (void)__hip_atomic_load(ctr, __ATOMIC_ACQUIRE, __HIP_MEMORY_SCOPE_AGENT);

    target += NWG;
    unsigned short* tmp = hcur; hcur = hnxt; hnxt = tmp;
  }
}

// ---------------------------------------------------------------------------
extern "C" void kernel_launch(void* const* d_in, const int* in_sizes, int n_in,
                              void* d_out, int out_size, void* d_ws, size_t ws_size,
                              hipStream_t stream) {
  const float* x  = (const float*)d_in[0];
  const float* h0 = (const float*)d_in[1];
  const float* W  = (const float*)d_in[2];
  const float* b  = (const float*)d_in[3];

  float* out   = (float*)d_out;                              // (B,S,DH)
  float* hlast = out + (size_t)B_ * S_ * DH;                 // (B,DH)

  char* ws = (char*)d_ws;
  unsigned short* wxT = (unsigned short*)(ws + WXT_OFF);
  unsigned short* whT = (unsigned short*)(ws + WHT_OFF);
  unsigned short* hb0 = (unsigned short*)(ws + HB0_OFF);
  unsigned short* hb1 = (unsigned short*)(ws + HB1_OFF);
  unsigned*       ctr = (unsigned*)(ws + CTR_OFF);

  rnn_init<<<256, 256, 0, stream>>>(W, h0, wxT, whT, hb0, ctr);

  dim3 grid(MTOT / BM, DH / BN);
  xproj_gemm<<<grid, 256, 0, stream>>>(x, wxT, b, out);

  rnn_scan<<<NWG, 256, 0, stream>>>(whT, out, hlast, hb0, hb1, ctr);
}
